// LSTM_82523501625822
// MI455X (gfx1250) — matrix operand order repaired and checked by reading, not compiled, not run on hardware
//
#include <hip/hip_runtime.h>

#define B_SZ 512
#define T_SZ 1024
#define H_SZ 128

typedef __attribute__((ext_vector_type(16))) _Float16 v16h;
typedef __attribute__((ext_vector_type(8)))  float    v8f;
typedef __attribute__((ext_vector_type(4)))  unsigned int u32x4;

union HFrag { u32x4 q[2]; v16h v; };

__device__ __forceinline__ float fast_sigmoid(float xv) {
    return __builtin_amdgcn_rcpf(1.0f + __expf(-xv));
}
__device__ __forceinline__ float fast_tanh(float xv) {
    float cx = fminf(fmaxf(xv, -15.0f), 15.0f);
    float e  = __expf(2.0f * cx);
    return (e - 1.0f) * __builtin_amdgcn_rcpf(e + 1.0f);
}

// Persistent LSTM kernel: one workgroup owns 16 batch rows for all 1024 steps.
// 8 waves / WG; wave w owns h-columns [16w,16w+16) and the matching i,f,g,o
// gate tiles. W_hh is held in VGPRs as f16 WMMA B-fragments for the whole scan.
__global__ __launch_bounds__(256, 1)
void lstm_persistent(const float* __restrict__ x,
                     const float* __restrict__ W_ih,
                     const float* __restrict__ W_hh,
                     const float* __restrict__ b_ih,
                     const float* __restrict__ b_hh,
                     const float* __restrict__ fc_w,
                     const float* __restrict__ fc_b,
                     float* __restrict__ out)
{
    // h state: 16 rows x 128 cols f16, row stride 136 halves (272B):
    // 16B-aligned rows, 4-bank stride across rows -> conflict-free b128 loads.
    __shared__ _Float16 hbuf[16 * 136];

    const int tid  = threadIdx.x;
    const int wave = tid >> 5;
    const int lane = tid & 31;
    const int lhi  = lane >> 4;   // 0 or 1 (half-wave)
    const int llo  = lane & 15;
    const int rowbase = blockIdx.x * 16;

    // ---- one-time: load W_hh as f16 B-fragments + input-weight scalars ----
    // gates[m,n] = sum_k h[m,k]*W_hh[n,k]  => B[k,n] = W_hh[n,k]
    // B-fragment lane layout: n = tile_base + (lane&15);
    // lanes 0-15 hold K = kb*32 + 0..15, lanes 16-31 hold K = kb*32 + 16..31.
    v16h  Bf[4][4];                 // [gate][k-block], 128 VGPRs/lane
    float w0[4], w1[4], bias[4];
#pragma unroll
    for (int g = 0; g < 4; ++g) {
        const int n = g * H_SZ + wave * 16 + llo;   // gate-space column
        w0[g]   = W_ih[n * 2 + 0];
        w1[g]   = W_ih[n * 2 + 1];
        bias[g] = b_ih[n] + b_hh[n];
#pragma unroll
        for (int kb = 0; kb < 4; ++kb) {
            const float* wr = W_hh + (size_t)n * H_SZ + kb * 32 + lhi * 16;
            v16h bv;
#pragma unroll
            for (int j = 0; j < 16; ++j) bv[j] = (_Float16)wr[j];
            Bf[g][kb] = bv;
        }
    }

    // zero h state
    for (int i = tid; i < 16 * 136; i += 256) hbuf[i] = (_Float16)0.0f;
    __syncthreads();

    // per-lane c state: rows M = v + 8*lhi, column = wave*16 + llo
    float c[8];
#pragma unroll
    for (int v = 0; v < 8; ++v) c[v] = 0.0f;

    const int hcol = wave * 16 + llo;

    // ---- sequential scan over T ----
    for (int t = 0; t < T_SZ; ++t) {
        // input contribution for this step (I == 2), rows owned by this lane
        float x0v[8], x1v[8];
#pragma unroll
        for (int v = 0; v < 8; ++v) {
            const int gm = rowbase + v + 8 * lhi;
            const float2 xv = *(const float2*)(x + ((size_t)gm * T_SZ + t) * 2);
            x0v[v] = xv.x; x1v[v] = xv.y;
        }
        if (t + 8 < T_SZ)  // pull x ahead through the cache hierarchy
            __builtin_prefetch(x + ((size_t)(rowbase + 8 * lhi) * T_SZ + (t + 8)) * 2, 0, 0);

        // pre-activation accumulators (C-matrix layout: row M = v+8*lhi, col = llo)
        v8f acc[4];
#pragma unroll
        for (int g = 0; g < 4; ++g)
#pragma unroll
            for (int v = 0; v < 8; ++v)
                acc[g][v] = fmaf(x0v[v], w0[g], fmaf(x1v[v], w1[g], bias[g]));

        // recurrent GEMM: 4 K-blocks x 4 gate tiles = 16 WMMAs, A reused 4x
#pragma unroll
        for (int kb = 0; kb < 4; ++kb) {
            HFrag a;
            const int arow = llo * 136;
            const int k0   = kb * 32 + lhi * 16;          // A layout contiguous, matches B
            a.q[0] = *(const u32x4*)&hbuf[arow + k0];      // K = k0 .. k0+7
            a.q[1] = *(const u32x4*)&hbuf[arow + k0 + 8];  // K = k0+8 .. k0+15
#pragma unroll
            for (int g = 0; g < 4; ++g)
                acc[g] = __builtin_amdgcn_wmma_f32_16x16x32_f16(
                    false, a.v, false, Bf[g][kb], (short)0, acc[g], false, false);
        }
        __syncthreads();   // everyone done reading h(t-1) from LDS

        // pointwise gates: i,f,g,o ; update c and write h(t)
#pragma unroll
        for (int v = 0; v < 8; ++v) {
            const float iv = fast_sigmoid(acc[0][v]);
            const float fv = fast_sigmoid(acc[1][v]);
            const float gv = fast_tanh   (acc[2][v]);
            const float ov = fast_sigmoid(acc[3][v]);
            const float cv = fmaf(fv, c[v], iv * gv);
            c[v] = cv;
            hbuf[(v + 8 * lhi) * 136 + hcol] = (_Float16)(ov * fast_tanh(cv));
        }
        __syncthreads();   // h(t) visible to all waves
    }

    // ---- epilogue: out[b] = relu(h_n[b]) . fc_w + fc_b ----
    if (tid < 16) {
        float s = fc_b[0];
        for (int k = 0; k < H_SZ; ++k) {
            const float hv = (float)hbuf[tid * 136 + k];
            s = fmaf(fmaxf(hv, 0.0f), fc_w[k], s);
        }
        out[rowbase + tid] = s;   // output shape (1, B, 1) flat
    }
}

extern "C" void kernel_launch(void* const* d_in, const int* in_sizes, int n_in,
                              void* d_out, int out_size, void* d_ws, size_t ws_size,
                              hipStream_t stream) {
    const float* x    = (const float*)d_in[0];
    const float* W_ih = (const float*)d_in[1];
    const float* W_hh = (const float*)d_in[2];
    const float* b_ih = (const float*)d_in[3];
    const float* b_hh = (const float*)d_in[4];
    const float* fc_w = (const float*)d_in[5];
    const float* fc_b = (const float*)d_in[6];
    float* out = (float*)d_out;

    dim3 grid(B_SZ / 16);   // 32 workgroups, one per 16-row batch tile
    dim3 block(256);        // 8 wave32s
    hipLaunchKernelGGL(lstm_persistent, grid, block, 0, stream,
                       x, W_ih, W_hh, b_ih, b_hh, fc_w, fc_b, out);
}
